// HCNet_68925635166929
// MI455X (gfx1250) — matrix operand order repaired and checked by reading, not compiled, not run on hardware
//
#include <hip/hip_runtime.h>

// ---------------- problem constants ----------------
#define D_    2048
#define N_    4
#define L_    4
#define MTOK  4096      // B*T = 2*2048

typedef __bf16 bf16;
typedef __bf16 v16bf __attribute__((ext_vector_type(16)));
typedef __bf16 v8bf  __attribute__((ext_vector_type(8)));
typedef float  v8f   __attribute__((ext_vector_type(8)));

// ---------------- CDNA5 async global->LDS copy (ASYNCcnt-tracked) ----------------
__device__ __forceinline__ void async_b128(unsigned lds_off, const void* gaddr) {
  // per-lane: LDS[lds_off] = MEM[gaddr], 16 bytes, tracked with ASYNCcnt
  asm volatile("global_load_async_to_lds_b128 %0, %1, off"
               :: "v"(lds_off), "v"(gaddr) : "memory");
}
__device__ __forceinline__ void wait_async0() {
  asm volatile("s_wait_asynccnt 0" ::: "memory");
}

// ---------------- helpers ----------------
__device__ __forceinline__ float block_reduce(float v, float* sred) {
  // wave32 reduce then cross-wave via LDS (8 waves / 256 threads)
  #pragma unroll
  for (int off = 16; off > 0; off >>= 1) v += __shfl_down(v, off, 32);
  const int tid = threadIdx.x;
  if ((tid & 31) == 0) sred[tid >> 5] = v;
  __syncthreads();
  float r = 0.f;
  #pragma unroll
  for (int w = 0; w < 8; ++w) r += sred[w];
  __syncthreads();
  return r;
}

// ---------------- init: H[t][n][d] = x[t][d] ----------------
__global__ __launch_bounds__(256) void init_kernel(const float* __restrict__ x,
                                                   float* __restrict__ H) {
  size_t idx = (size_t)blockIdx.x * 256 + threadIdx.x;   // over MTOK*N_*D_
  int d = (int)(idx % D_);
  size_t t = idx / ((size_t)N_ * D_);
  H[idx] = x[t * D_ + d];
}

// ---------------- weight transpose + f32->bf16: Wt[e][d] = W[d][e] ----------------
__global__ __launch_bounds__(256) void wconv_kernel(const float* __restrict__ W,
                                                    bf16* __restrict__ Wt) {
  __shared__ float tile[32][33];
  const int bx = blockIdx.x * 32;            // e-base
  const int by = blockIdx.y * 32;            // d-base
  const int tx = threadIdx.x & 31, ty = threadIdx.x >> 5;
  #pragma unroll
  for (int j = 0; j < 32; j += 8)
    tile[ty + j][tx] = W[(size_t)(by + ty + j) * D_ + bx + tx];
  __syncthreads();
  #pragma unroll
  for (int j = 0; j < 32; j += 8)
    Wt[(size_t)(bx + ty + j) * D_ + by + tx] = (bf16)tile[tx][ty + j];
}

// ---------------- width connections: one workgroup per token ----------------
__global__ __launch_bounds__(256) void width_kernel(
    float* __restrict__ H, bf16* __restrict__ hln,
    const float* __restrict__ Am_, const float* __restrict__ Ar_,
    const float* __restrict__ Wm_, const float* __restrict__ Wr_,
    const float* __restrict__ sa_, const float* __restrict__ dg_,
    const float* __restrict__ ng_, const float* __restrict__ nb_) {
  __shared__ float sred[8];
  const int t = blockIdx.x;
  const int tid = threadIdx.x;
  float* Hrow = H + (size_t)t * N_ * D_;

  float hold[N_][8], hn[N_][8], dg[8];
  #pragma unroll
  for (int j = 0; j < 8; ++j) dg[j] = dg_[tid + j * 256];
  #pragma unroll
  for (int n = 0; n < N_; ++n)
    #pragma unroll
    for (int j = 0; j < 8; ++j) hold[n][j] = Hrow[n * D_ + tid + j * 256];

  const float sa = sa_[0];
  // dynamic LN per n (no bias)
  #pragma unroll
  for (int n = 0; n < N_; ++n) {
    float s = 0.f, q = 0.f;
    #pragma unroll
    for (int j = 0; j < 8; ++j) { s += hold[n][j]; q += hold[n][j] * hold[n][j]; }
    float S  = block_reduce(s, sred);
    float Q  = block_reduce(q, sred);
    float mu = S * (1.0f / D_);
    float var = Q * (1.0f / D_) - mu * mu;
    float rs = rsqrtf(var + 1e-5f);
    #pragma unroll
    for (int j = 0; j < 8; ++j) hn[n][j] = (hold[n][j] - mu) * rs * dg[j];
  }

  // Am (Nx1) and Ar (NxN) dots over D
  float amp[N_] = {0.f, 0.f, 0.f, 0.f};
  float arp[N_][N_] = {};
  #pragma unroll
  for (int j = 0; j < 8; ++j) {
    const int d = tid + j * 256;
    const float wm = Wm_[d];
    const float4 wr = *(const float4*)(Wr_ + (size_t)d * N_);
    #pragma unroll
    for (int n = 0; n < N_; ++n) {
      const float xv = hn[n][j];
      amp[n] += xv * wm;
      arp[n][0] += xv * wr.x; arp[n][1] += xv * wr.y;
      arp[n][2] += xv * wr.z; arp[n][3] += xv * wr.w;
    }
  }
  float am[N_], ar[N_][N_];
  #pragma unroll
  for (int n = 0; n < N_; ++n) {
    am[n] = Am_[n] + sa * tanhf(block_reduce(amp[n], sred));
    #pragma unroll
    for (int m = 0; m < N_; ++m)
      ar[n][m] = Ar_[n * N_ + m] + sa * tanhf(block_reduce(arp[n][m], sred));
  }

  // h = H^T am ; Hnew[m] = H^T ar[:,m]  (in place: hold is in registers)
  float hv[8];
  #pragma unroll
  for (int j = 0; j < 8; ++j) {
    float h = 0.f;
    #pragma unroll
    for (int n = 0; n < N_; ++n) h += hold[n][j] * am[n];
    hv[j] = h;
    #pragma unroll
    for (int m = 0; m < N_; ++m) {
      float v = 0.f;
      #pragma unroll
      for (int n = 0; n < N_; ++n) v += hold[n][j] * ar[n][m];
      Hrow[m * D_ + tid + j * 256] = v;
    }
  }

  // pre-norm LN -> bf16 GEMM input
  float s = 0.f, q = 0.f;
  #pragma unroll
  for (int j = 0; j < 8; ++j) { s += hv[j]; q += hv[j] * hv[j]; }
  float S  = block_reduce(s, sred);
  float Q  = block_reduce(q, sred);
  float mu = S * (1.0f / D_);
  float rs = rsqrtf(Q * (1.0f / D_) - mu * mu + 1e-5f);
  #pragma unroll
  for (int j = 0; j < 8; ++j) {
    const int d = tid + j * 256;
    hln[(size_t)t * D_ + d] = (bf16)((hv[j] - mu) * rs * ng_[d] + nb_[d]);
  }
}

// ---------------- bf16 WMMA GEMM: C[M,Nd] = A[M,K] * W[K,Nd] + bias ----------------
// A row-major bf16 (hln); Bt[n][k] = W[k][n] row-major bf16 (transposed weights).
// Tiles staged with CDNA5 async global->LDS copies (ASYNCcnt), double buffered.
#define BM 128
#define BN 128
#define BK 64
#define NSTAGE (D_ / BK)    // 32

__global__ __launch_bounds__(256) void gemm_bf16(
    const bf16* __restrict__ A, const bf16* __restrict__ Bt,
    const float* __restrict__ bias, float* __restrict__ C) {
  __shared__ __align__(32) bf16 sA[2][BM][BK];
  __shared__ __align__(32) bf16 sB[2][BN][BK];

  const int tid  = threadIdx.x;
  const int gm0  = blockIdx.y * BM;
  const int gn0  = blockIdx.x * BN;
  const int lane = tid & 31;
  const int wave = tid >> 5;
  const int wm   = wave >> 1;        // 0..3 -> 32-row band
  const int wn   = wave & 1;         // 0..1 -> 64-col band
  const int lr   = lane & 15;
  const int hi   = lane >> 4;

  const bf16* Ag = A  + (size_t)gm0 * D_;
  const bf16* Bg = Bt + (size_t)gn0 * D_;

  // issue one stage: 128x64 bf16 per matrix = 1024 x 16B chunks -> 4 per thread
  auto issue = [&](int kb, int buf) {
    #pragma unroll
    for (int c = 0; c < 4; ++c) {
      const int idx = tid + c * 256;
      const int row = idx >> 3, seg = idx & 7;      // 8 chunks per 64-elem row
      async_b128((unsigned)(uintptr_t)&sA[buf][row][seg * 8],
                 Ag + (size_t)row * D_ + kb * BK + seg * 8);
      async_b128((unsigned)(uintptr_t)&sB[buf][row][seg * 8],
                 Bg + (size_t)row * D_ + kb * BK + seg * 8);
    }
  };

  v8f acc[2][4];
  #pragma unroll
  for (int tm = 0; tm < 2; ++tm)
    #pragma unroll
    for (int tn = 0; tn < 4; ++tn) acc[tm][tn] = (v8f){0.f,0.f,0.f,0.f,0.f,0.f,0.f,0.f};

  issue(0, 0);

  for (int kb = 0; kb < NSTAGE; ++kb) {
    const int buf = kb & 1;
    wait_async0();                 // this wave's stage-kb tiles landed in LDS
    __syncthreads();               // all waves' tiles landed; prev reads of buf^1 done
    if (kb + 1 < NSTAGE) issue(kb + 1, buf ^ 1);   // overlap copy with compute

    #pragma unroll
    for (int ks = 0; ks < 2; ++ks) {               // two K=32 sub-steps
      // B fragments: lane col lr holds K = ks*32 + hi*16 .. +15 contiguous
      v16bf bfr[4];
      #pragma unroll
      for (int tn = 0; tn < 4; ++tn)
        bfr[tn] = *(const v16bf*)(&sB[buf][wn * 64 + tn * 16 + lr][ks * 32 + hi * 16]);

      #pragma unroll
      for (int tm = 0; tm < 2; ++tm) {
        // A fragment: lane row lr, K = hi*16 .. +15 contiguous (within sub-step)
        const bf16* p = &sA[buf][wm * 32 + tm * 16 + lr][ks * 32];
        v8bf alo = *(const v8bf*)(p + hi * 16);
        v8bf ahi = *(const v8bf*)(p + hi * 16 + 8);
        v16bf af = __builtin_shufflevector(alo, ahi,
                     0,1,2,3,4,5,6,7,8,9,10,11,12,13,14,15);
        #pragma unroll
        for (int tn = 0; tn < 4; ++tn)
          acc[tm][tn] = __builtin_amdgcn_wmma_f32_16x16x32_bf16(
              false, af, false, bfr[tn], (short)0, acc[tm][tn], false, false);
      }
    }
  }

  // epilogue: C layout -> lane group hi: M = v + 8*hi, N = lr
  #pragma unroll
  for (int tm = 0; tm < 2; ++tm) {
    const int row0 = gm0 + wm * 32 + tm * 16 + hi * 8;
    #pragma unroll
    for (int tn = 0; tn < 4; ++tn) {
      const int col = gn0 + wn * 64 + tn * 16 + lr;
      const float b = bias[col];
      #pragma unroll
      for (int v = 0; v < 8; ++v)
        C[(size_t)(row0 + v) * D_ + col] = acc[tm][tn][v] + b;
    }
  }
}

// ---------------- depth connections: one workgroup per token ----------------
__global__ __launch_bounds__(256) void depth_kernel(
    const float* __restrict__ hout, float* __restrict__ H,
    const float* __restrict__ Bp_, const float* __restrict__ Wb_,
    const float* __restrict__ sb_, const float* __restrict__ dg_) {
  __shared__ float sred[8];
  const int t = blockIdx.x;
  const int tid = threadIdx.x;

  float hv[8];
  #pragma unroll
  for (int j = 0; j < 8; ++j) hv[j] = hout[(size_t)t * D_ + tid + j * 256];

  float s = 0.f, q = 0.f;
  #pragma unroll
  for (int j = 0; j < 8; ++j) { s += hv[j]; q += hv[j] * hv[j]; }
  float S  = block_reduce(s, sred);
  float Q  = block_reduce(q, sred);
  float mu = S * (1.0f / D_);
  float rs = rsqrtf(Q * (1.0f / D_) - mu * mu + 1e-5f);

  float bvp[N_] = {0.f, 0.f, 0.f, 0.f};
  #pragma unroll
  for (int j = 0; j < 8; ++j) {
    const int d = tid + j * 256;
    const float xv = (hv[j] - mu) * rs * dg_[d];
    const float4 wb = *(const float4*)(Wb_ + (size_t)d * N_);
    bvp[0] += xv * wb.x; bvp[1] += xv * wb.y;
    bvp[2] += xv * wb.z; bvp[3] += xv * wb.w;
  }
  const float sb = sb_[0];
  float bv[N_];
  #pragma unroll
  for (int n = 0; n < N_; ++n)
    bv[n] = Bp_[n] + sb * tanhf(block_reduce(bvp[n], sred));

  float* Hrow = H + (size_t)t * N_ * D_;
  #pragma unroll
  for (int n = 0; n < N_; ++n)
    #pragma unroll
    for (int j = 0; j < 8; ++j)
      Hrow[n * D_ + tid + j * 256] += hv[j] * bv[n];
}

// ---------------- final: out[t][d] = sum_n H[t][n][d] ----------------
__global__ __launch_bounds__(256) void final_kernel(const float* __restrict__ H,
                                                    float* __restrict__ out) {
  size_t idx = (size_t)blockIdx.x * 256 + threadIdx.x;   // over MTOK*D_
  size_t t = idx / D_;
  int d = (int)(idx % D_);
  const float* p = H + t * (size_t)N_ * D_ + d;
  out[idx] = p[0] + p[D_] + p[2 * D_] + p[3 * D_];
}

// ---------------- host driver ----------------
extern "C" void kernel_launch(void* const* d_in, const int* in_sizes, int n_in,
                              void* d_out, int out_size, void* d_ws, size_t ws_size,
                              hipStream_t stream) {
  const float* x          = (const float*)d_in[0];
  const float* A_m        = (const float*)d_in[1];
  const float* A_r        = (const float*)d_in[2];
  const float* Bp         = (const float*)d_in[3];
  const float* W_m        = (const float*)d_in[4];
  const float* W_r        = (const float*)d_in[5];
  const float* W_b        = (const float*)d_in[6];
  const float* s_a        = (const float*)d_in[7];
  const float* s_b        = (const float*)d_in[8];
  const float* dyn_gamma  = (const float*)d_in[9];
  const float* norm_gamma = (const float*)d_in[10];
  const float* norm_beta  = (const float*)d_in[11];
  const float* Wblk       = (const float*)d_in[12];
  const float* bblk       = (const float*)d_in[13];
  float* out = (float*)d_out;

  char* ws = (char*)d_ws;
  float* Hcur = (float*)ws;  ws += (size_t)MTOK * N_ * D_ * sizeof(float);   // 128 MB
  bf16*  hln  = (bf16*)ws;   ws += (size_t)MTOK * D_ * sizeof(bf16);         // 16 MB
  float* hout = (float*)ws;  ws += (size_t)MTOK * D_ * sizeof(float);        // 32 MB
  bf16*  Wbf  = (bf16*)ws;                                                   // 8 MB

  init_kernel<<<(MTOK * N_ * D_) / 256, 256, 0, stream>>>(x, Hcur);

  for (int i = 0; i < L_; ++i) {
    wconv_kernel<<<dim3(D_ / 32, D_ / 32), 256, 0, stream>>>(
        Wblk + (size_t)i * D_ * D_, Wbf);
    width_kernel<<<MTOK, 256, 0, stream>>>(
        Hcur, hln,
        A_m + i * N_, A_r + i * N_ * N_,
        W_m + (size_t)i * D_, W_r + (size_t)i * D_ * N_,
        s_a + i, dyn_gamma + (size_t)i * D_,
        norm_gamma + (size_t)i * D_, norm_beta + (size_t)i * D_);
    gemm_bf16<<<dim3(D_ / BN, MTOK / BM), 256, 0, stream>>>(
        hln, Wbf, bblk + (size_t)i * D_, hout);
    depth_kernel<<<MTOK, 256, 0, stream>>>(
        hout, Hcur,
        Bp + i * N_, W_b + (size_t)i * D_ * N_,
        s_b + i, dyn_gamma + (size_t)i * D_);
  }

  final_kernel<<<(MTOK * D_) / 256, 256, 0, stream>>>(Hcur, out);
}
